// BaseMoELayer_28063316313058
// MI455X (gfx1250) — matrix operand order repaired and checked by reading, not compiled, not run on hardware
//
#include <hip/hip_runtime.h>
#include <hip/hip_bf16.h>
#include <math.h>

typedef __bf16 bf16;
typedef __attribute__((ext_vector_type(16))) __bf16 v16bf;
typedef __attribute__((ext_vector_type(8)))  float  v8f;

#define T_TOK 4096
#define D_IN  2048
#define N_EXP 8
#define H_DIM 704
#define TOPK  2
#define Y_ELEMS 8388608ull /* 2*2048*2048 */

#define BM 128
#define BN 64
#define BK 32
#define LDP (BK + 8)  /* LDS row stride in bf16: 80B, 16B-aligned, bank-skewed */
#define R_CAP 9216    /* 8192 routed rows + per-expert 128-pad */

__device__ __forceinline__ int ka_of(int el, int hi) {
  // contiguous K order (k = 16*hi + el), matching the B fragment reads
  return 16 * hi + el;
}

__device__ __forceinline__ v8f wmma_bf16(v16bf a, v16bf b, v8f c) {
  return __builtin_amdgcn_wmma_f32_16x16x32_bf16(false, a, false, b, (short)0, c,
                                                 false, false);
}

__device__ __forceinline__ unsigned pack_bf16(float lo, float hi) {
  union { bf16 h[2]; unsigned u; } p;
  p.h[0] = (bf16)lo;
  p.h[1] = (bf16)hi;
  return p.u;
}

// ---------------------------------------------------------------- utilities
__global__ void zero_f32_kernel(float* __restrict__ p, unsigned long long n) {
  unsigned long long i = (unsigned long long)blockIdx.x * blockDim.x + threadIdx.x;
  unsigned long long stride = (unsigned long long)gridDim.x * blockDim.x;
  for (; i < n; i += stride) p[i] = 0.0f;
}

__global__ void init_small_kernel(int* counts, int* cursors, float* importance) {
  int i = threadIdx.x;
  if (i < N_EXP) { counts[i] = 0; cursors[i] = 0; importance[i] = 0.0f; }
}

// ---------------------------------------------------------------- gate
__global__ __launch_bounds__(256)
void gate_kernel(const float* __restrict__ x, const float* __restrict__ Wg1,
                 const float* __restrict__ Wg2, int* __restrict__ top_i,
                 float* __restrict__ top_s, float* __restrict__ importance,
                 int* __restrict__ counts) {
  const int wave = threadIdx.x >> 5;
  const int lane = threadIdx.x & 31;
  const int t = blockIdx.x * 8 + wave;
  if (t >= T_TOK) return;
  const float* xr = x + (size_t)t * D_IN;
  float acc[N_EXP];
#pragma unroll
  for (int e = 0; e < N_EXP; ++e) acc[e] = 0.0f;
  for (int k = lane; k < D_IN; k += 32) {
    float xv = xr[k];
    const float4* w4 = (const float4*)(Wg1 + (size_t)k * N_EXP);
    float4 wa = w4[0], wb = w4[1];
    acc[0] += xv * wa.x; acc[1] += xv * wa.y;
    acc[2] += xv * wa.z; acc[3] += xv * wa.w;
    acc[4] += xv * wb.x; acc[5] += xv * wb.y;
    acc[6] += xv * wb.z; acc[7] += xv * wb.w;
  }
#pragma unroll
  for (int e = 0; e < N_EXP; ++e) {
    float v = acc[e];
#pragma unroll
    for (int m = 16; m >= 1; m >>= 1) v += __shfl_xor(v, m, 32);
    acc[e] = v;
  }
  if (lane == 0) {
    float a[N_EXP], lg[N_EXP];
#pragma unroll
    for (int e = 0; e < N_EXP; ++e) a[e] = tanhf(acc[e]);
#pragma unroll
    for (int j = 0; j < N_EXP; ++j) {
      float s = 0.0f;
#pragma unroll
      for (int e = 0; e < N_EXP; ++e) s += a[e] * Wg2[e * N_EXP + j];
      lg[j] = s;
    }
    int i0 = 0;
#pragma unroll
    for (int j = 1; j < N_EXP; ++j) if (lg[j] > lg[i0]) i0 = j;
    int i1 = (i0 == 0) ? 1 : 0;
#pragma unroll
    for (int j = 0; j < N_EXP; ++j)
      if (j != i0 && lg[j] > lg[i1]) i1 = j;
    float m = lg[i0];
    float e0 = __expf(lg[i0] - m), e1 = __expf(lg[i1] - m);
    float inv = 1.0f / (e0 + e1);
    float s0 = e0 * inv, s1 = e1 * inv;
    top_i[t * 2 + 0] = i0; top_i[t * 2 + 1] = i1;
    top_s[t * 2 + 0] = s0; top_s[t * 2 + 1] = s1;
    atomicAdd(&importance[i0], s0);
    atomicAdd(&importance[i1], s1);
    atomicAdd(&counts[i0], 1);
    atomicAdd(&counts[i1], 1);
  }
}

// ------------------------------------------------- offsets + balance loss
__global__ void finalize_kernel(const int* __restrict__ counts,
                                const float* __restrict__ importance,
                                int* __restrict__ offsets,
                                float* __restrict__ loss_out) {
  if (threadIdx.x == 0) {
    int off = 0;
    float mi = 0.0f, ml = 0.0f;
    for (int e = 0; e < N_EXP; ++e) {
      offsets[e] = off;
      off += ((counts[e] + BM - 1) / BM) * BM;
      mi += importance[e];
      ml += (float)counts[e];
    }
    mi *= (1.0f / N_EXP); ml *= (1.0f / N_EXP);
    float vi = 0.0f, vl = 0.0f;
    for (int e = 0; e < N_EXP; ++e) {
      float di = importance[e] - mi; vi += di * di;
      float dl = (float)counts[e] - ml; vl += dl * dl;
    }
    vi *= (1.0f / (N_EXP - 1)); vl *= (1.0f / (N_EXP - 1));  // ddof=1
    *loss_out = 0.01f * (vi / (mi * mi + 1e-10f) + vl / (ml * ml + 1e-10f));
  }
}

// ---------------------------------------------------------------- scatter
__global__ void scatter_kernel(const int* __restrict__ top_i,
                               const float* __restrict__ top_s,
                               const int* __restrict__ offsets,
                               int* __restrict__ cursors, int* __restrict__ rtok,
                               float* __restrict__ rsc) {
  int t = blockIdx.x * blockDim.x + threadIdx.x;
  if (t >= T_TOK) return;
  for (int k = 0; k < TOPK; ++k) {
    int e = top_i[t * 2 + k];
    int pos = atomicAdd(&cursors[e], 1);
    int r = offsets[e] + pos;
    rtok[r] = t;
    rsc[r] = top_s[t * 2 + k];
  }
}

// -------------------------------------------- GEMM1: fused gate+up (SwiGLU)
__global__ __launch_bounds__(256)
void expert_up_kernel(const float* __restrict__ x, const float* __restrict__ Wg,
                      const float* __restrict__ Wu, const int* __restrict__ counts,
                      const int* __restrict__ offsets, const int* __restrict__ rtok,
                      const float* __restrict__ rsc, bf16* __restrict__ hbuf) {
  const int e = blockIdx.z;
  const int n_e = counts[e];
  const int rowBase = blockIdx.x * BM;
  if (rowBase >= n_e) return;
  const int colBase = blockIdx.y * BN;
  const int off = offsets[e];

  __shared__ __align__(16) bf16 As[BM][LDP];
  __shared__ __align__(16) bf16 BgT[BN][LDP];
  __shared__ __align__(16) bf16 BuT[BN][LDP];

  const float* __restrict__ Wge = Wg + (size_t)e * D_IN * H_DIM;
  const float* __restrict__ Wue = Wu + (size_t)e * D_IN * H_DIM;

  const int tid = threadIdx.x;
  const int wave = tid >> 5;
  const int lane = tid & 31;
  const int nlane = lane & 15;
  const int hi = lane >> 4;

  // A staging: 2 threads/row, 16 f32 (4x b128) each, rows gathered via token ids
  const int sa_row = tid >> 1;
  const int sa_half = (tid & 1) * 16;
  const int sa_grow = rowBase + sa_row;
  const float* sa_src = nullptr;
  if (sa_grow < n_e) {
    int tok = rtok[off + sa_grow];
    sa_src = x + (size_t)tok * D_IN + sa_half;
  }
  // B staging: each thread owns a 2(K) x 4(N) micro-tile, stored transposed
  const int sb_k = (tid >> 4) * 2;   // 0,2,..,30
  const int sb_n = (tid & 15) * 4;   // 0,4,..,60

  v8f zero8 = {0.f, 0.f, 0.f, 0.f, 0.f, 0.f, 0.f, 0.f};
  v8f accg[4] = {zero8, zero8, zero8, zero8};
  v8f accu[4] = {zero8, zero8, zero8, zero8};

  float4 fA[4], fG[2], fU[2];
  auto preload = [&](int kk) {
    if (sa_src) {
      const float4* s4 = (const float4*)(sa_src + kk);
#pragma unroll
      for (int i = 0; i < 4; ++i) fA[i] = s4[i];
    }
    const float* g0 = Wge + (size_t)(kk + sb_k) * H_DIM + colBase + sb_n;
    const float* u0 = Wue + (size_t)(kk + sb_k) * H_DIM + colBase + sb_n;
    fG[0] = *(const float4*)g0;
    fG[1] = *(const float4*)(g0 + H_DIM);
    fU[0] = *(const float4*)u0;
    fU[1] = *(const float4*)(u0 + H_DIM);
  };

  preload(0);
  for (int kk = 0; kk < D_IN; kk += BK) {
    if (kk) __syncthreads();  // previous tile's LDS reads complete
    // publish staged registers to LDS (bf16)
    if (sa_src) {
      union { unsigned u[8]; uint4 q[2]; } pk;
#pragma unroll
      for (int i = 0; i < 4; ++i) {
        pk.u[2 * i + 0] = pack_bf16(fA[i].x, fA[i].y);
        pk.u[2 * i + 1] = pack_bf16(fA[i].z, fA[i].w);
      }
      uint4* dst = (uint4*)&As[sa_row][sa_half];
      dst[0] = pk.q[0];
      dst[1] = pk.q[1];
    } else if (kk == 0) {
      uint4 z; z.x = z.y = z.z = z.w = 0u;
      uint4* dst = (uint4*)&As[sa_row][sa_half];
      dst[0] = z;
      dst[1] = z;
    }
    *(unsigned*)&BgT[sb_n + 0][sb_k] = pack_bf16(fG[0].x, fG[1].x);
    *(unsigned*)&BgT[sb_n + 1][sb_k] = pack_bf16(fG[0].y, fG[1].y);
    *(unsigned*)&BgT[sb_n + 2][sb_k] = pack_bf16(fG[0].z, fG[1].z);
    *(unsigned*)&BgT[sb_n + 3][sb_k] = pack_bf16(fG[0].w, fG[1].w);
    *(unsigned*)&BuT[sb_n + 0][sb_k] = pack_bf16(fU[0].x, fU[1].x);
    *(unsigned*)&BuT[sb_n + 1][sb_k] = pack_bf16(fU[0].y, fU[1].y);
    *(unsigned*)&BuT[sb_n + 2][sb_k] = pack_bf16(fU[0].z, fU[1].z);
    *(unsigned*)&BuT[sb_n + 3][sb_k] = pack_bf16(fU[0].w, fU[1].w);
    __syncthreads();

    // software pipeline: kick off next tile's global loads before compute
    if (kk + BK < D_IN) preload(kk + BK);

    v16bf afrag;
    const int arow = wave * 16 + nlane;
#pragma unroll
    for (int el = 0; el < 16; ++el) afrag[el] = As[arow][ka_of(el, hi)];

#pragma unroll
    for (int ct = 0; ct < 4; ++ct) {
      v16bf bg, bu;
      const bf16* bgrow = &BgT[ct * 16 + nlane][hi * 16];
      const bf16* burow = &BuT[ct * 16 + nlane][hi * 16];
#pragma unroll
      for (int el = 0; el < 16; ++el) { bg[el] = bgrow[el]; bu[el] = burow[el]; }
      accg[ct] = wmma_bf16(afrag, bg, accg[ct]);
      accu[ct] = wmma_bf16(afrag, bu, accu[ct]);
    }
  }

  // epilogue: h = silu(g)*u*score  (score folded in before down-proj)
  float sc[8];
  int grows[8];
#pragma unroll
  for (int i = 0; i < 8; ++i) {
    int g = rowBase + wave * 16 + i + 8 * hi;
    grows[i] = g;
    sc[i] = (g < n_e) ? rsc[off + g] : 0.0f;
  }
#pragma unroll
  for (int ct = 0; ct < 4; ++ct) {
#pragma unroll
    for (int i = 0; i < 8; ++i) {
      int g = grows[i];
      if (g < n_e) {
        float gv = accg[ct][i];
        float uv = accu[ct][i];
        float hv = (gv / (1.0f + __expf(-gv))) * uv * sc[i];
        hbuf[(size_t)(off + g) * H_DIM + colBase + ct * 16 + nlane] = (bf16)hv;
      }
    }
  }
}

// ------------------------------------------------------ GEMM2: down-proj
__global__ __launch_bounds__(256)
void expert_down_kernel(const bf16* __restrict__ hbuf, const float* __restrict__ Wd,
                        const int* __restrict__ counts,
                        const int* __restrict__ offsets,
                        const int* __restrict__ rtok, float* __restrict__ y) {
  const int e = blockIdx.z;
  const int n_e = counts[e];
  const int rowBase = blockIdx.x * BM;
  if (rowBase >= n_e) return;
  const int colBase = blockIdx.y * BN;
  const int off = offsets[e];

  __shared__ __align__(16) bf16 As[BM][LDP];
  __shared__ __align__(16) bf16 BT[BN][LDP];

  const float* __restrict__ Wde = Wd + (size_t)e * H_DIM * D_IN;

  const int tid = threadIdx.x;
  const int wave = tid >> 5;
  const int lane = tid & 31;
  const int nlane = lane & 15;
  const int hi = lane >> 4;

  const int sa_row = tid >> 1;
  const int sa_half = (tid & 1) * 16;
  const int sa_grow = rowBase + sa_row;
  const bf16* sa_src =
      (sa_grow < n_e) ? (hbuf + (size_t)(off + sa_grow) * H_DIM + sa_half) : nullptr;

  const int sb_k = (tid >> 4) * 2;
  const int sb_n = (tid & 15) * 4;

  v8f zero8 = {0.f, 0.f, 0.f, 0.f, 0.f, 0.f, 0.f, 0.f};
  v8f acc[4] = {zero8, zero8, zero8, zero8};

  uint4 rA[2];
  float4 fB[2];
  auto preload = [&](int kk) {
    if (sa_src) {
      const uint4* s = (const uint4*)(sa_src + kk);  // bf16 data, direct copy
      rA[0] = s[0];
      rA[1] = s[1];
    }
    const float* b0 = Wde + (size_t)(kk + sb_k) * D_IN + colBase + sb_n;
    fB[0] = *(const float4*)b0;
    fB[1] = *(const float4*)(b0 + D_IN);
  };

  preload(0);
  for (int kk = 0; kk < H_DIM; kk += BK) {
    if (kk) __syncthreads();
    if (sa_src) {
      uint4* dst = (uint4*)&As[sa_row][sa_half];
      dst[0] = rA[0];
      dst[1] = rA[1];
    } else if (kk == 0) {
      uint4 z; z.x = z.y = z.z = z.w = 0u;
      uint4* dst = (uint4*)&As[sa_row][sa_half];
      dst[0] = z;
      dst[1] = z;
    }
    *(unsigned*)&BT[sb_n + 0][sb_k] = pack_bf16(fB[0].x, fB[1].x);
    *(unsigned*)&BT[sb_n + 1][sb_k] = pack_bf16(fB[0].y, fB[1].y);
    *(unsigned*)&BT[sb_n + 2][sb_k] = pack_bf16(fB[0].z, fB[1].z);
    *(unsigned*)&BT[sb_n + 3][sb_k] = pack_bf16(fB[0].w, fB[1].w);
    __syncthreads();

    if (kk + BK < H_DIM) preload(kk + BK);

    v16bf afrag;
    const int arow = wave * 16 + nlane;
#pragma unroll
    for (int el = 0; el < 16; ++el) afrag[el] = As[arow][ka_of(el, hi)];
#pragma unroll
    for (int ct = 0; ct < 4; ++ct) {
      v16bf b;
      const bf16* brow = &BT[ct * 16 + nlane][hi * 16];
#pragma unroll
      for (int el = 0; el < 16; ++el) b[el] = brow[el];
      acc[ct] = wmma_bf16(afrag, b, acc[ct]);
    }
  }

  int toks[8];
#pragma unroll
  for (int i = 0; i < 8; ++i) {
    int g = rowBase + wave * 16 + i + 8 * hi;
    toks[i] = (g < n_e) ? rtok[off + g] : -1;
  }
#pragma unroll
  for (int ct = 0; ct < 4; ++ct) {
#pragma unroll
    for (int i = 0; i < 8; ++i) {
      if (toks[i] >= 0)
        atomicAdd(&y[(size_t)toks[i] * D_IN + colBase + ct * 16 + nlane],
                  acc[ct][i]);
    }
  }
}

// ---------------------------------------------------------------- launcher
extern "C" void kernel_launch(void* const* d_in, const int* in_sizes, int n_in,
                              void* d_out, int out_size, void* d_ws, size_t ws_size,
                              hipStream_t stream) {
  (void)in_sizes; (void)n_in; (void)out_size; (void)ws_size;
  const float* x     = (const float*)d_in[0];
  const float* Wg1   = (const float*)d_in[1];
  const float* Wg2   = (const float*)d_in[2];
  const float* Wgate = (const float*)d_in[3];
  const float* Wup   = (const float*)d_in[4];
  const float* Wdown = (const float*)d_in[5];
  // d_in[6] = num_selects (== 2, baked into TOPK)

  float* y = (float*)d_out;
  float* loss = y + Y_ELEMS;

  char* ws = (char*)d_ws;
  int*   counts     = (int*)(ws + 0);
  int*   cursors    = (int*)(ws + 64);
  int*   offsets    = (int*)(ws + 128);
  float* importance = (float*)(ws + 192);
  size_t o = 256;
  int*   top_i = (int*)(ws + o);   o += (size_t)T_TOK * TOPK * 4;
  float* top_s = (float*)(ws + o); o += (size_t)T_TOK * TOPK * 4;
  int*   rtok  = (int*)(ws + o);   o += (size_t)R_CAP * 4;
  float* rsc   = (float*)(ws + o); o += (size_t)R_CAP * 4;
  o = (o + 255) & ~(size_t)255;
  bf16*  hbuf  = (bf16*)(ws + o);  // R_CAP * H_DIM bf16 (~13 MB)

  init_small_kernel<<<1, 64, 0, stream>>>(counts, cursors, importance);
  zero_f32_kernel<<<2048, 256, 0, stream>>>(y, Y_ELEMS);
  gate_kernel<<<T_TOK / 8, 256, 0, stream>>>(x, Wg1, Wg2, top_i, top_s,
                                             importance, counts);
  finalize_kernel<<<1, 32, 0, stream>>>(counts, importance, offsets, loss);
  scatter_kernel<<<T_TOK / 256, 256, 0, stream>>>(top_i, top_s, offsets, cursors,
                                                  rtok, rsc);
  expert_up_kernel<<<dim3(T_TOK / BM, H_DIM / BN, N_EXP), 256, 0, stream>>>(
      x, Wgate, Wup, counts, offsets, rtok, rsc, hbuf);
  expert_down_kernel<<<dim3(T_TOK / BM, D_IN / BN, N_EXP), 256, 0, stream>>>(
      hbuf, Wdown, counts, offsets, rtok, y);
}
